// MoETopX_71571335020841
// MI455X (gfx1250) — compile-verified, matrix operand order checked by reading, not run on hardware
//
#include <hip/hip_runtime.h>

typedef __attribute__((ext_vector_type(16))) __bf16 v16bf;
typedef __attribute__((ext_vector_type(8)))  float  v8f;
typedef unsigned short ushort_t;

#define N_TOKENS  8192
#define D_MODEL   2048
#define OUT_DIM   2048
#define N_EXPERTS 8
#define MAX_PER_E N_TOKENS
#define GW        16                    // waves per block (expert kernel)
#define COLS_PER_WAVE (OUT_DIM / GW)    // 128
#define NT        (COLS_PER_WAVE / 16)  // 8 WMMA col-subtiles per wave
#define MROWS     32                    // row tile (2 x 16-row A fragments)
#define KSTEP     32
#define ROW_TILES (MAX_PER_E / MROWS)   // 256 row tiles max per expert

// workspace layout
#define WS_COUNTS   0
#define WS_BIDX     256
#define WS_BW       (WS_BIDX + N_EXPERTS * MAX_PER_E * 4)
#define WS_XB       (1u << 20)                                   // bf16 X [N][D], 32 MB
#define WS_WT       (WS_XB + N_TOKENS * D_MODEL * 2)             // bf16 Wt [E][O][D], 64 MB

// ---- f32 -> bf16 (round-to-nearest-even) ----
__device__ __forceinline__ ushort_t f2bf(float f) {
  unsigned int u = __float_as_uint(f);
  u = (u + 0x7FFFu + ((u >> 16) & 1u)) >> 16;
  return (ushort_t)u;
}

// ---------------- kernel: zero output + bucket counters ----------------
__global__ void moe_zero_kernel(float4* __restrict__ out4, int n4, int* __restrict__ counts) {
  int i = blockIdx.x * blockDim.x + threadIdx.x;
  if (i < n4) out4[i] = make_float4(0.f, 0.f, 0.f, 0.f);
  if (blockIdx.x == 0 && threadIdx.x < N_EXPERTS) counts[threadIdx.x] = 0;
}

// ---------------- kernel: convert inputs f32 -> bf16 (8 elems / thread) ----------------
__global__ void moe_cvt_x_kernel(const float* __restrict__ x, ushort_t* __restrict__ xb) {
  int i = blockIdx.x * blockDim.x + threadIdx.x;   // one uint4 (8 bf16) per thread
  const float4* xi = (const float4*)x + (size_t)i * 2;
  float4 a = xi[0], b = xi[1];
  union { ushort_t s[8]; uint4 q; } o;
  o.s[0] = f2bf(a.x); o.s[1] = f2bf(a.y); o.s[2] = f2bf(a.z); o.s[3] = f2bf(a.w);
  o.s[4] = f2bf(b.x); o.s[5] = f2bf(b.y); o.s[6] = f2bf(b.z); o.s[7] = f2bf(b.w);
  ((uint4*)xb)[i] = o.q;
}

// ---------------- kernel: convert + transpose We[e][k][n] f32 -> Wt[e][n][k] bf16 ----------------
__global__ void moe_cvt_w_kernel(const float* __restrict__ We, ushort_t* __restrict__ Wt) {
  __shared__ ushort_t tile[64][65];   // [k][n], padded
  int e   = blockIdx.x >> 10;         // / (32*32)
  int rem = blockIdx.x & 1023;
  int k0  = (rem >> 5) << 6;          // 64-row K tile
  int n0  = (rem & 31) << 6;          // 64-col N tile
  int tid = threadIdx.x;

  const float* src = We + ((size_t)e * D_MODEL) * OUT_DIM;
#pragma unroll
  for (int i = 0; i < 4; ++i) {       // 1024 float4 / 256 threads
    int vec = tid + i * 256;
    int n4  = vec & 15;               // 16 float4 per 64-wide row
    int kk  = vec >> 4;               // 0..63
    float4 v = *(const float4*)(src + (size_t)(k0 + kk) * OUT_DIM + n0 + n4 * 4);
    tile[kk][n4 * 4 + 0] = f2bf(v.x);
    tile[kk][n4 * 4 + 1] = f2bf(v.y);
    tile[kk][n4 * 4 + 2] = f2bf(v.z);
    tile[kk][n4 * 4 + 3] = f2bf(v.w);
  }
  __syncthreads();
  int n   = tid >> 2;                 // 64 rows, 4 threads per output row
  int kb  = (tid & 3) << 4;           // 16 k each
  union { ushort_t s[16]; uint4 q[2]; } o;
#pragma unroll
  for (int j = 0; j < 16; ++j) o.s[j] = tile[kb + j][n];
  uint4* dst = (uint4*)(Wt + ((size_t)e * OUT_DIM + n0 + n) * D_MODEL + k0 + kb);
  dst[0] = o.q[0];
  dst[1] = o.q[1];
}

// ---------------- kernel: gate + top-2 + double softmax + routing ----------------
__global__ void moe_gate_kernel(const float* __restrict__ x,
                                const float* __restrict__ Wg,
                                const float* __restrict__ bg,
                                int* __restrict__ counts,
                                int* __restrict__ bucket_idx,
                                float* __restrict__ bucket_w) {
  int wave  = threadIdx.x >> 5;
  int lane  = threadIdx.x & 31;
  int token = blockIdx.x * (blockDim.x >> 5) + wave;
  if (token >= N_TOKENS) return;

  const float* xr = x + (size_t)token * D_MODEL;
  float acc[N_EXPERTS];
#pragma unroll
  for (int j = 0; j < N_EXPERTS; ++j) acc[j] = 0.f;

  for (int k = lane; k < D_MODEL; k += 32) {
    float xv = xr[k];
    const float4* wr = (const float4*)(Wg + (size_t)k * N_EXPERTS);
    float4 w0 = wr[0], w1 = wr[1];
    acc[0] += xv * w0.x; acc[1] += xv * w0.y; acc[2] += xv * w0.z; acc[3] += xv * w0.w;
    acc[4] += xv * w1.x; acc[5] += xv * w1.y; acc[6] += xv * w1.z; acc[7] += xv * w1.w;
  }
#pragma unroll
  for (int j = 0; j < N_EXPERTS; ++j)
    for (int m = 16; m >= 1; m >>= 1) acc[j] += __shfl_xor(acc[j], m, 32);

  if (lane == 0) {
    float v[N_EXPERTS];
#pragma unroll
    for (int j = 0; j < N_EXPERTS; ++j) v[j] = acc[j] + bg[j];
    int i1 = 0; float b1 = v[0];
    for (int j = 1; j < N_EXPERTS; ++j) if (v[j] > b1) { b1 = v[j]; i1 = j; }
    int i2 = -1; float b2 = -3.4e38f;
    for (int j = 0; j < N_EXPERTS; ++j) if (j != i1 && v[j] > b2) { b2 = v[j]; i2 = j; }
    float e2 = __expf(b2 - b1);
    float s  = 1.f + e2;
    float w1 = 1.f / s, w2 = e2 / s;
    float f2 = __expf(w2 - w1);
    float s2 = 1.f + f2;
    float W1 = 1.f / s2, W2 = f2 / s2;

    int p1 = atomicAdd(&counts[i1], 1);
    bucket_idx[i1 * MAX_PER_E + p1] = token;
    bucket_w  [i1 * MAX_PER_E + p1] = W1;
    int p2 = atomicAdd(&counts[i2], 1);
    bucket_idx[i2 * MAX_PER_E + p2] = token;
    bucket_w  [i2 * MAX_PER_E + p2] = W2;
  }
}

// ---------------- kernel: gathered bf16 GEMM + bias + ReLU + softmax + scatter ----------------
__global__ __launch_bounds__(512) void moe_expert_kernel(
    const ushort_t* __restrict__ Xb,   // bf16 [N][D]
    const ushort_t* __restrict__ Wt,   // bf16 [E][O][D] (n-major)
    const float*    __restrict__ be,
    const int*      __restrict__ counts,
    const int*      __restrict__ bucket_idx,
    const float*    __restrict__ bucket_w,
    float* __restrict__ out) {
  __shared__ int   tokS[MROWS];
  __shared__ float coefS[MROWS];
  __shared__ float red_max[GW][MROWS];
  __shared__ float red_sum[GW][MROWS];

  int e    = blockIdx.x >> 8;          // / ROW_TILES
  int rt   = blockIdx.x & (ROW_TILES - 1);
  int cnt  = counts[e];
  int row0 = rt * MROWS;
  if (row0 >= cnt) return;

  if (threadIdx.x < MROWS) {
    int r = row0 + threadIdx.x;
    bool valid = (r < cnt);
    tokS[threadIdx.x]  = bucket_idx[e * MAX_PER_E + (valid ? r : row0)];
    coefS[threadIdx.x] = valid ? bucket_w[e * MAX_PER_E + r] : 0.f;
  }
  __syncthreads();

  int wave = threadIdx.x >> 5;
  int lane = threadIdx.x & 31;
  int l16  = lane & 15;
  int half = (lane >> 4) << 4;         // 0 or 16 : K-half for A/B frags
  int rowb = half >> 1;                // 0 or 8  : C/D row base
  int colbase = wave * COLS_PER_WAVE;

  v8f acc[2][NT] = {};                 // 2 row groups x 8 col tiles

  const ushort_t* xrow0 = Xb + (size_t)tokS[l16]      * D_MODEL + half;
  const ushort_t* xrow1 = Xb + (size_t)tokS[16 + l16] * D_MODEL + half;
  // B base: col (colbase + l16), tile t adds 16 rows = 32768 ushorts (folds to imm offset)
  const ushort_t* bbase = Wt + ((size_t)e * OUT_DIM + colbase + l16) * D_MODEL + half;

  for (int kb = 0; kb < D_MODEL; kb += KSTEP) {
    union { uint4 q[2]; v16bf v; } a0, a1;
    { const uint4* p = (const uint4*)(xrow0 + kb); a0.q[0] = p[0]; a0.q[1] = p[1]; }
    { const uint4* p = (const uint4*)(xrow1 + kb); a1.q[0] = p[0]; a1.q[1] = p[1]; }
    const uint4* bp = (const uint4*)(bbase + kb);
#pragma unroll
    for (int t = 0; t < NT; ++t) {
      union { uint4 q[2]; v16bf v; } bfrag;
      bfrag.q[0] = bp[t * 4096];       // t*16 rows * 2048 ushorts = 4096 uint4
      bfrag.q[1] = bp[t * 4096 + 1];
      acc[0][t] = __builtin_amdgcn_wmma_f32_16x16x32_bf16(
          false, a0.v, false, bfrag.v, (short)0, acc[0][t], false, false);
      acc[1][t] = __builtin_amdgcn_wmma_f32_16x16x32_bf16(
          false, a1.v, false, bfrag.v, (short)0, acc[1][t], false, false);
    }
  }

  // ---- bias + ReLU ----
#pragma unroll
  for (int t = 0; t < NT; ++t) {
    float bev = be[e * OUT_DIM + colbase + t * 16 + l16];
#pragma unroll
    for (int g = 0; g < 2; ++g)
#pragma unroll
      for (int j = 0; j < 8; ++j) acc[g][t][j] = fmaxf(acc[g][t][j] + bev, 0.f);
  }

  // ---- row max ----
  float pm[2][8];
#pragma unroll
  for (int g = 0; g < 2; ++g)
#pragma unroll
    for (int j = 0; j < 8; ++j) {
      float m = acc[g][0][j];
#pragma unroll
      for (int t = 1; t < NT; ++t) m = fmaxf(m, acc[g][t][j]);
      for (int msk = 8; msk >= 1; msk >>= 1) m = fmaxf(m, __shfl_xor(m, msk, 32));
      pm[g][j] = m;
    }
  if (l16 == 0) {
#pragma unroll
    for (int g = 0; g < 2; ++g)
#pragma unroll
      for (int j = 0; j < 8; ++j) red_max[wave][g * 16 + rowb + j] = pm[g][j];
  }
  __syncthreads();
  float rm[2][8];
#pragma unroll
  for (int g = 0; g < 2; ++g)
#pragma unroll
    for (int j = 0; j < 8; ++j) {
      float m = red_max[0][g * 16 + rowb + j];
      for (int w2 = 1; w2 < GW; ++w2) m = fmaxf(m, red_max[w2][g * 16 + rowb + j]);
      rm[g][j] = m;
    }

  // ---- exp + row sum ----
  float ps[2][8];
#pragma unroll
  for (int g = 0; g < 2; ++g)
#pragma unroll
    for (int j = 0; j < 8; ++j) ps[g][j] = 0.f;
#pragma unroll
  for (int t = 0; t < NT; ++t)
#pragma unroll
    for (int g = 0; g < 2; ++g)
#pragma unroll
      for (int j = 0; j < 8; ++j) {
        float v = __expf(acc[g][t][j] - rm[g][j]);
        acc[g][t][j] = v;
        ps[g][j] += v;
      }
#pragma unroll
  for (int g = 0; g < 2; ++g)
#pragma unroll
    for (int j = 0; j < 8; ++j)
      for (int msk = 8; msk >= 1; msk >>= 1) ps[g][j] += __shfl_xor(ps[g][j], msk, 32);
  if (l16 == 0) {
#pragma unroll
    for (int g = 0; g < 2; ++g)
#pragma unroll
      for (int j = 0; j < 8; ++j) red_sum[wave][g * 16 + rowb + j] = ps[g][j];
  }
  __syncthreads();
  float rinv[2][8];
#pragma unroll
  for (int g = 0; g < 2; ++g)
#pragma unroll
    for (int j = 0; j < 8; ++j) {
      float s = red_sum[0][g * 16 + rowb + j];
      for (int w2 = 1; w2 < GW; ++w2) s += red_sum[w2][g * 16 + rowb + j];
      rinv[g][j] = 1.f / s;
    }

  // ---- scale by routing coef, scatter-accumulate ----
#pragma unroll
  for (int t = 0; t < NT; ++t) {
    int n = colbase + t * 16 + l16;
#pragma unroll
    for (int g = 0; g < 2; ++g)
#pragma unroll
      for (int j = 0; j < 8; ++j) {
        int r = g * 16 + rowb + j;
        float val = acc[g][t][j] * rinv[g][j] * coefS[r];
        unsafeAtomicAdd(&out[(size_t)tokS[r] * OUT_DIM + n], val);
      }
  }
}

// ---------------- launch ----------------
extern "C" void kernel_launch(void* const* d_in, const int* in_sizes, int n_in,
                              void* d_out, int out_size, void* d_ws, size_t ws_size,
                              hipStream_t stream) {
  const float* x  = (const float*)d_in[0];
  const float* We = (const float*)d_in[1];
  const float* be = (const float*)d_in[2];
  const float* Wg = (const float*)d_in[3];
  const float* bg = (const float*)d_in[4];
  float* out = (float*)d_out;

  char* ws = (char*)d_ws;
  int*      counts     = (int*)(ws + WS_COUNTS);
  int*      bucket_idx = (int*)(ws + WS_BIDX);
  float*    bucket_w   = (float*)(ws + WS_BW);
  ushort_t* Xb         = (ushort_t*)(ws + WS_XB);
  ushort_t* Wt         = (ushort_t*)(ws + WS_WT);

  int n4 = (N_TOKENS * OUT_DIM) / 4;
  moe_zero_kernel<<<(n4 + 255) / 256, 256, 0, stream>>>((float4*)out, n4, counts);
  moe_cvt_x_kernel<<<(N_TOKENS * D_MODEL / 8) / 256, 256, 0, stream>>>(x, Xb);
  moe_cvt_w_kernel<<<N_EXPERTS * 32 * 32, 256, 0, stream>>>(We, Wt);
  moe_gate_kernel<<<N_TOKENS / 8, 256, 0, stream>>>(x, Wg, bg, counts, bucket_idx, bucket_w);
  moe_expert_kernel<<<N_EXPERTS * ROW_TILES, 512, 0, stream>>>(
      Xb, Wt, be, counts, bucket_idx, bucket_w, out);
}
